// RetinaNet_64974265254173
// MI455X (gfx1250) — matrix operand order repaired and checked by reading, not compiled, not run on hardware
//
#include <hip/hip_runtime.h>
#include <hip/hip_bf16.h>
#include <math.h>

typedef __attribute__((ext_vector_type(16))) __bf16 v16bf;
typedef __attribute__((ext_vector_type(8)))  float  v8f;
typedef unsigned int u32x4 __attribute__((ext_vector_type(4)));
typedef int          i32x4 __attribute__((ext_vector_type(4)));
typedef int          i32x8 __attribute__((ext_vector_type(8)));

#define NC    91
#define MAXC  5000
#define ATOT  182403
#define KTOT  2304      // 9 * 256

__device__ __constant__ int   c_lvlOff[5] = {0, 136800, 171000, 179550, 181773};
__device__ __constant__ int   c_lvlW[5]   = {152, 76, 38, 19, 10};
__device__ __constant__ float c_stride[5] = {8.f, 16.f, 32.f, 64.f, 128.f};
__device__ __constant__ float c_astart[5] = {32.f, 64.f, 128.f, 256.f, 512.f};

__device__ inline unsigned short f2bf(float f) {
  unsigned int u = __float_as_uint(f);
  return (unsigned short)((u + 0x7FFFu + ((u >> 16) & 1u)) >> 16);
}

// ---------------- layout converters ----------------

// OIHW f32 -> wgtT[co][ (ky*3+kx)*Cin + ci ] bf16  (K contiguous per output channel)
__global__ void k_wcvt(const float* __restrict__ w, unsigned short* __restrict__ out,
                       int Cin, int Cout) {
  long idx = (long)blockIdx.x * blockDim.x + threadIdx.x;
  long tot = (long)Cout * Cin * 9;
  if (idx >= tot) return;
  int co  = (int)(idx / (Cin * 9));
  int rem = (int)(idx - (long)co * (Cin * 9));
  int ci  = rem / 9;
  int kk  = rem - ci * 9;
  out[(long)co * (Cin * 9) + kk * Cin + ci] = f2bf(w[idx]);
}

// NCHW f32 -> NHWC bf16
__global__ void k_nchw2nhwc(const float* __restrict__ in, unsigned short* __restrict__ out,
                            int Bn, int C, int H, int W) {
  long idx = (long)blockIdx.x * blockDim.x + threadIdx.x;
  long tot = (long)Bn * C * H * W;
  if (idx >= tot) return;
  int c = (int)(idx % C); long r = idx / C;
  int x = (int)(r % W);   r /= W;
  int y = (int)(r % H);   int b = (int)(r / H);
  out[idx] = f2bf(in[(((long)b * C + c) * H + y) * W + x]);
}

__global__ void k_zero_counts(int* candCount, int Bn) {
  int i = threadIdx.x;
  if (i < Bn) candCount[i] = 0;
}

// ---------------- implicit-GEMM 3x3 conv via WMMA bf16 + TDM weight staging ----------
// Block tile 64(M) x 64(N); 8 wave32s as 4(M) x 2(N); each wave owns 16x32 (2 WMMA accs).
// Per 3x3 tap: TDM pulls the 64x256 weight panel into LDS while threads stage the 64x256
// activation panel; then 8 unrolled K=32 slices -> 16 WMMAs between barrier pairs.
// mode 0: relu -> bf16 NHWC out (hidden layers, Cout==256)
// mode 1: f32 out at of32[b*ostrideB + obase + (y*W+x)*Cout + col]  (reg final)
// mode 2: fused sigmoid + threshold + candidate append              (cls final)
__global__ __launch_bounds__(256, 2)
void k_conv_gemm(const unsigned short* __restrict__ in,    // NHWC bf16 [B,H,W,256]
                 const unsigned short* __restrict__ wgtT,  // [Cout, 2304] bf16
                 const float* __restrict__ bias,           // [Cout]
                 int Bn, int H, int W, int Cout, int mode,
                 unsigned short* __restrict__ obf,
                 float* __restrict__ of32, long obase, long ostrideB,
                 float* __restrict__ candScore, int* __restrict__ candAnchor,
                 int* __restrict__ candCls, int* __restrict__ candCount,
                 int lvlOff) {
  __shared__ __align__(16) unsigned short s_lds[2 * 64 * 256];   // 64 KB
  unsigned short* Al = s_lds;              // [64 rows][256 k]   activations
  unsigned short* Bl = s_lds + 64 * 256;   // [64 n   ][256 k]   weights (via TDM)

  const int Cin  = 256;
  const int M    = Bn * H * W;
  const int mblk = blockIdx.x * 64;
  const int nblk = blockIdx.y * 64;
  const int t    = threadIdx.x;
  const int lane = t & 31;
  const int wave = t >> 5;
  const int wm   = wave & 3;
  const int wn   = wave >> 2;
  const int hi   = (lane >> 4) & 1;
  const int ln   = lane & 15;

  // A staging: thread -> (row, 64-element segment)
  const int ar   = t >> 2;
  const int aseg = (t & 3) * 64;
  const int am   = mblk + ar;
  int ab = 0, ay = 0, ax = 0;
  const bool aval = (am < M);
  if (aval) { ab = am / (H * W); int rm = am - ab * (H * W); ay = rm / W; ax = rm - ay * W; }

  v8f acc[2];
  acc[0] = (v8f){0.f,0.f,0.f,0.f,0.f,0.f,0.f,0.f};
  acc[1] = acc[0];

  for (int tap = 0; tap < 9; ++tap) {
    const int ky = tap / 3 - 1;
    const int kx = tap - (tap / 3) * 3 - 1;
    __syncthreads();                         // previous tap fully consumed

    // ---- TDM: 64(n) x 256(k) weight panel -> LDS (OOB rows auto zero-filled) ----
    if (wave == 0) {
      unsigned long long ga =
          (unsigned long long)(const void*)(wgtT + (size_t)nblk * KTOT + (size_t)tap * 256);
      u32x4 g0;
      g0.x = 1u;                                                    // count=1
      g0.y = 64u * 256u * 2u;                                       // LDS byte offset of B panel
      g0.z = (unsigned)(ga & 0xFFFFFFFFull);
      g0.w = (unsigned)((ga >> 32) & 0x1FFFFFFull) | 0x80000000u;   // type=2 ("image")
      const unsigned td0 = (unsigned)(KTOT - tap * 256);            // remaining k in tensor
      const unsigned td1 = (unsigned)(Cout - nblk);                 // remaining n rows
      i32x8 g1;
      g1[0] = (int)(1u << 16);                                      // data_size = 2 bytes
      g1[1] = (int)((td0 & 0xFFFFu) << 16);                         // tensor_dim0 lo
      g1[2] = (int)((td0 >> 16) | ((td1 & 0xFFFFu) << 16));         // dim0 hi | dim1 lo
      g1[3] = (int)((td1 >> 16) | (256u << 16));                    // dim1 hi | tile_dim0=256
      g1[4] = (int)64u;                                             // tile_dim1=64, tile_dim2=0
      g1[5] = (int)(unsigned)KTOT;                                  // tensor_dim0_stride
      g1[6] = 0; g1[7] = 0;
      i32x4 gz = {0, 0, 0, 0};
#if defined(__clang_major__) && (__clang_major__ >= 23)
      i32x8 gz8 = {0, 0, 0, 0, 0, 0, 0, 0};
      __builtin_amdgcn_tensor_load_to_lds(g0, g1, gz, gz, gz8, 0);
#else
      __builtin_amdgcn_tensor_load_to_lds(g0, g1, gz, gz, 0);
#endif
    }

    // ---- A panel: shifted 64x256 activation window, zero padded ----
    {
      const int yy = ay + ky, xx = ax + kx;
      const bool v = aval && (unsigned)yy < (unsigned)H && (unsigned)xx < (unsigned)W;
      const unsigned short* src = in + (((long)(ab * H + yy)) * W + xx) * Cin + aseg;
      if (v) __builtin_prefetch(src + Cin, 0, 0);
#pragma unroll
      for (int j = 0; j < 8; ++j) {
        uint4 d = make_uint4(0u, 0u, 0u, 0u);
        if (v) d = *(const uint4*)(src + j * 8);
        *(uint4*)&Al[ar * 256 + aseg + j * 8] = d;
      }
    }

    if (wave == 0) __builtin_amdgcn_s_wait_tensorcnt(0);
    __syncthreads();                        // panels visible to all waves

    // ---- 8 K-slices x 2 WMMA tiles per wave ----
    const int arow  = (wm * 16 + ln) * 256;
    const int brow0 = (wn * 32 + ln) * 256;
    const int brow1 = (wn * 32 + 16 + ln) * 256;
#pragma unroll
    for (int ks = 0; ks < 8; ++ks) {
      union { v16bf v; uint4 q[2]; } af, b0, b1;
      af.q[0] = *(const uint4*)&Al[arow + ks * 32 + hi * 16];
      af.q[1] = *(const uint4*)&Al[arow + ks * 32 + hi * 16 + 8];
      b0.q[0] = *(const uint4*)&Bl[brow0 + ks * 32 + hi * 16];
      b0.q[1] = *(const uint4*)&Bl[brow0 + ks * 32 + hi * 16 + 8];
      b1.q[0] = *(const uint4*)&Bl[brow1 + ks * 32 + hi * 16];
      b1.q[1] = *(const uint4*)&Bl[brow1 + ks * 32 + hi * 16 + 8];
      acc[0] = __builtin_amdgcn_wmma_f32_16x16x32_bf16(false, af.v, false, b0.v,
                                                       (short)0, acc[0], false, false);
      acc[1] = __builtin_amdgcn_wmma_f32_16x16x32_bf16(false, af.v, false, b1.v,
                                                       (short)0, acc[1], false, false);
    }
  }

  // ---- epilogue ----
  const int hw = H * W;
#pragma unroll
  for (int sub = 0; sub < 2; ++sub) {
    const int col = nblk + wn * 32 + sub * 16 + ln;
    const float bsv = (col < Cout) ? bias[col] : 0.f;
#pragma unroll
    for (int i = 0; i < 8; ++i) {
      const int m = mblk + wm * 16 + hi * 8 + i;    // C/D layout: VGPR i -> M=i (+8 hi lanes)
      if (m >= M || col >= Cout) continue;
      float v = acc[sub][i] + bsv;
      const int b  = m / hw;
      const int rm = m - b * hw;                    // y*W + x
      if (mode == 0) {
        v = fmaxf(v, 0.f);
        obf[(long)m * Cout + col] = f2bf(v);
      } else if (mode == 1) {
        of32[(long)b * ostrideB + obase + (long)rm * Cout + col] = v;
      } else {
        const float sc = 1.f / (1.f + __expf(-v));
        if (sc > 0.05f) {
          const int a   = col / NC;
          const int cls = col - a * NC;
          const int slot = atomicAdd(&candCount[b], 1);
          if (slot < MAXC) {
            candScore [b * MAXC + slot] = sc;
            candAnchor[b * MAXC + slot] = lvlOff + rm * 9 + a;
            candCls   [b * MAXC + slot] = cls;
          }
        }
      }
    }
  }
}

// ---------------- box decode (priors computed analytically) ----------------
__global__ void k_decode(const float* __restrict__ regAll,
                         const int* __restrict__ candAnchor, const int* __restrict__ candCls,
                         const int* __restrict__ candCount,
                         float* __restrict__ candBox, int Bn) {
  const int b = blockIdx.y;
  const int i = blockIdx.x * blockDim.x + threadIdx.x;
  int n = candCount[b]; if (n > MAXC) n = MAXC;
  if (i >= n) return;
  const int A = candAnchor[b * MAXC + i];
  int lvl = 0;
  if (A >= c_lvlOff[1]) lvl = 1;
  if (A >= c_lvlOff[2]) lvl = 2;
  if (A >= c_lvlOff[3]) lvl = 3;
  if (A >= c_lvlOff[4]) lvl = 4;
  const int rem  = A - c_lvlOff[lvl];
  const int cell = rem / 9;
  const int a    = rem - cell * 9;
  const int Wl   = c_lvlW[lvl];
  const int y = cell / Wl, x = cell - (cell / Wl) * Wl;
  const int si = a / 3, ri = a - (a / 3) * 3;
  const float s  = (si == 0) ? 1.f : ((si == 1) ? 1.2599210498948732f : 1.5874010519681994f);
  const float rr = (ri == 0) ? 2.f : ((ri == 1) ? 1.f : 0.5f);
  const float size = c_astart[lvl] * s;
  const float wa = roundf(size * sqrtf(rr));
  const float ha = roundf(size / sqrtf(rr));
  const float st = c_stride[lvl];
  const float pcx = (x + 0.5f) * st, pcy = (y + 0.5f) * st;
  const float* rg = regAll + ((long)b * ATOT + A) * 4;
  const float cx = rg[0] * wa + pcx;
  const float cy = rg[1] * ha + pcy;
  const float CLIP = 4.135166556742356f;          // ln(1000/16)
  const float ww = __expf(fminf(rg[2], CLIP)) * wa;
  const float hh = __expf(fminf(rg[3], CLIP)) * ha;
  float x1 = cx - 0.5f * ww, y1 = cy - 0.5f * hh;
  float x2 = cx + 0.5f * ww, y2 = cy + 0.5f * hh;
  x1 = fminf(fmaxf(x1, 0.f), 1216.f); x2 = fminf(fmaxf(x2, 0.f), 1216.f);
  y1 = fminf(fmaxf(y1, 0.f),  800.f); y2 = fminf(fmaxf(y2, 0.f),  800.f);
  float* ob = candBox + ((long)b * MAXC + i) * 4;
  ob[0] = x1; ob[1] = y1; ob[2] = x2; ob[3] = y2;
}

// ---------------- class-aware iterative NMS (one block per image) ----------------
__global__ void k_nms(const float* __restrict__ candBox, float* __restrict__ candScore,
                      const int* __restrict__ candCls, const int* __restrict__ candCount,
                      float* __restrict__ outBoxes, float* __restrict__ outScores,
                      float* __restrict__ outCls) {
  __shared__ float sVal[256];
  __shared__ int   sIdx[256];
  const int b = blockIdx.x;
  const int t = threadIdx.x;
  int n = candCount[b]; if (n > MAXC) n = MAXC;
  const float* bx = candBox + (long)b * MAXC * 4;
  float* sc       = candScore + (long)b * MAXC;
  const int* cl   = candCls + (long)b * MAXC;

  for (int it = 0; it < 300; it++) {
    float best = -1e30f; int bi = -1;
    for (int i = t; i < n; i += 256) { float v = sc[i]; if (v > best) { best = v; bi = i; } }
    sVal[t] = best; sIdx[t] = bi;
    __syncthreads();
    for (int s2 = 128; s2 > 0; s2 >>= 1) {
      if (t < s2 && sVal[t + s2] > sVal[t]) { sVal[t] = sVal[t + s2]; sIdx[t] = sIdx[t + s2]; }
      __syncthreads();
    }
    const int   i0 = sIdx[0];
    const float s0 = sVal[0];
    __syncthreads();
    const bool valid = (i0 >= 0) && (s0 > 0.05f);
    if (t == 0) {
      const long o = (long)b * 300 + it;
      if (valid) {
        outBoxes[o * 4 + 0] = bx[i0 * 4 + 0]; outBoxes[o * 4 + 1] = bx[i0 * 4 + 1];
        outBoxes[o * 4 + 2] = bx[i0 * 4 + 2]; outBoxes[o * 4 + 3] = bx[i0 * 4 + 3];
        outScores[o] = s0; outCls[o] = (float)cl[i0];
      } else {
        outBoxes[o * 4 + 0] = 0.f; outBoxes[o * 4 + 1] = 0.f;
        outBoxes[o * 4 + 2] = 0.f; outBoxes[o * 4 + 3] = 0.f;
        outScores[o] = 0.f; outCls[o] = 0.f;
      }
    }
    if (valid) {
      const float cOff = cl[i0] * 1217.f;
      const float X1 = bx[i0 * 4 + 0] + cOff, Y1 = bx[i0 * 4 + 1] + cOff;
      const float X2 = bx[i0 * 4 + 2] + cOff, Y2 = bx[i0 * 4 + 3] + cOff;
      const float a1 = (X2 - X1) * (Y2 - Y1);
      for (int i = t; i < n; i += 256) {
        const float c2 = cl[i] * 1217.f;
        const float u1 = bx[i * 4 + 0] + c2, v1 = bx[i * 4 + 1] + c2;
        const float u2 = bx[i * 4 + 2] + c2, v2 = bx[i * 4 + 3] + c2;
        const float iw = fmaxf(fminf(X2, u2) - fmaxf(X1, u1), 0.f);
        const float ih = fmaxf(fminf(Y2, v2) - fmaxf(Y1, v1), 0.f);
        const float inter = iw * ih;
        const float a2 = (u2 - u1) * (v2 - v1);
        if (inter / (a1 + a2 - inter + 1e-9f) > 0.5f) sc[i] = -2.f;
      }
    }
    __syncthreads();
  }
}

// ---------------- host orchestration ----------------
extern "C" void kernel_launch(void* const* d_in, const int* in_sizes, int n_in,
                              void* d_out, int out_size, void* d_ws, size_t ws_size,
                              hipStream_t stream) {
  (void)in_sizes; (void)n_in; (void)out_size; (void)ws_size;
  const float* feat[5];
  for (int i = 0; i < 5; i++) feat[i] = (const float*)d_in[i];
  const float* cls_cw = (const float*)d_in[5];
  const float* cls_cb = (const float*)d_in[6];
  const float* cls_fw = (const float*)d_in[7];
  const float* cls_fb = (const float*)d_in[8];
  const float* reg_cw = (const float*)d_in[9];
  const float* reg_cb = (const float*)d_in[10];
  const float* reg_fw = (const float*)d_in[11];
  const float* reg_fb = (const float*)d_in[12];

  const int Bn = 2, C = 256;
  static const int LH[5]   = {100, 50, 25, 13, 7};
  static const int LW[5]   = {152, 76, 38, 19, 10};
  static const int LOFF[5] = {0, 136800, 171000, 179550, 181773};

  char* ws = (char*)d_ws;
  size_t off = 0;
  auto take = [&](size_t bytes) -> char* {
    char* p = ws + off; off += (bytes + 255) & ~(size_t)255; return p;
  };

  unsigned short* wCC = (unsigned short*)take(4ull * KTOT * 256 * 2);
  unsigned short* wCF = (unsigned short*)take((size_t)KTOT * 819 * 2);
  unsigned short* wRC = (unsigned short*)take(4ull * KTOT * 256 * 2);
  unsigned short* wRF = (unsigned short*)take((size_t)KTOT * 36 * 2);
  const size_t featElems = (size_t)Bn * 100 * 152 * 256;        // largest level
  unsigned short* actIn = (unsigned short*)take(featElems * 2);
  unsigned short* act0  = (unsigned short*)take(featElems * 2);
  unsigned short* act1  = (unsigned short*)take(featElems * 2);
  float* regAll     = (float*)take((size_t)Bn * ATOT * 4 * 4);
  int*   candCount  = (int*)take((size_t)Bn * 4);
  float* candScore  = (float*)take((size_t)Bn * MAXC * 4);
  int*   candAnchor = (int*)take((size_t)Bn * MAXC * 4);
  int*   candCls    = (int*)take((size_t)Bn * MAXC * 4);
  float* candBox    = (float*)take((size_t)Bn * MAXC * 16);

  // weight repack f32 -> bf16 [Cout][2304] (once per launch)
  for (int k = 0; k < 4; k++) {
    const long tot = (long)256 * 256 * 9;
    k_wcvt<<<dim3((unsigned)((tot + 255) / 256)), 256, 0, stream>>>(
        cls_cw + (size_t)k * tot, wCC + (size_t)k * KTOT * 256, 256, 256);
    k_wcvt<<<dim3((unsigned)((tot + 255) / 256)), 256, 0, stream>>>(
        reg_cw + (size_t)k * tot, wRC + (size_t)k * KTOT * 256, 256, 256);
  }
  { const long tot = (long)819 * 256 * 9;
    k_wcvt<<<dim3((unsigned)((tot + 255) / 256)), 256, 0, stream>>>(cls_fw, wCF, 256, 819); }
  { const long tot = (long)36 * 256 * 9;
    k_wcvt<<<dim3((unsigned)((tot + 255) / 256)), 256, 0, stream>>>(reg_fw, wRF, 256, 36); }
  k_zero_counts<<<1, 32, 0, stream>>>(candCount, Bn);

  auto conv = [&](const unsigned short* in, const unsigned short* w, const float* bias,
                  int H, int W, int Cout, int mode,
                  unsigned short* obf, float* of32, long obase, long ostrideB, int lvlOff) {
    const int M = Bn * H * W;
    dim3 g((unsigned)((M + 63) / 64), (unsigned)((Cout + 63) / 64));
    k_conv_gemm<<<g, 256, 0, stream>>>(in, w, bias, Bn, H, W, Cout, mode,
                                       obf, of32, obase, ostrideB,
                                       candScore, candAnchor, candCls, candCount, lvlOff);
  };

  for (int l = 0; l < 5; l++) {
    const int H = LH[l], W = LW[l];
    const long tot = (long)Bn * C * H * W;
    k_nchw2nhwc<<<dim3((unsigned)((tot + 255) / 256)), 256, 0, stream>>>(
        feat[l], actIn, Bn, C, H, W);
    // cls head: 4x conv+relu then fused sigmoid/threshold final
    conv(actIn, wCC + 0ull * KTOT * 256, cls_cb + 0,   H, W, 256, 0, act0, nullptr, 0, 0, 0);
    conv(act0,  wCC + 1ull * KTOT * 256, cls_cb + 256, H, W, 256, 0, act1, nullptr, 0, 0, 0);
    conv(act1,  wCC + 2ull * KTOT * 256, cls_cb + 512, H, W, 256, 0, act0, nullptr, 0, 0, 0);
    conv(act0,  wCC + 3ull * KTOT * 256, cls_cb + 768, H, W, 256, 0, act1, nullptr, 0, 0, 0);
    conv(act1,  wCF, cls_fb, H, W, 819, 2, nullptr, nullptr, 0, 0, LOFF[l]);
    // reg head: 4x conv+relu then f32 store into regAll [B, A_total, 4]
    conv(actIn, wRC + 0ull * KTOT * 256, reg_cb + 0,   H, W, 256, 0, act0, nullptr, 0, 0, 0);
    conv(act0,  wRC + 1ull * KTOT * 256, reg_cb + 256, H, W, 256, 0, act1, nullptr, 0, 0, 0);
    conv(act1,  wRC + 2ull * KTOT * 256, reg_cb + 512, H, W, 256, 0, act0, nullptr, 0, 0, 0);
    conv(act0,  wRC + 3ull * KTOT * 256, reg_cb + 768, H, W, 256, 0, act1, nullptr, 0, 0, 0);
    conv(act1,  wRF, reg_fb, H, W, 36, 1, nullptr, regAll, (long)LOFF[l] * 4, (long)ATOT * 4, 0);
  }

  k_decode<<<dim3((MAXC + 255) / 256, Bn), 256, 0, stream>>>(
      regAll, candAnchor, candCls, candCount, candBox, Bn);

  float* outBoxes  = (float*)d_out;
  float* outScores = outBoxes + (size_t)Bn * 300 * 4;
  float* outCls    = outScores + (size_t)Bn * 300;
  k_nms<<<Bn, 256, 0, stream>>>(candBox, candScore, candCls, candCount,
                                outBoxes, outScores, outCls);
}
